// HybridCsiNetQuantumAutoencoder_70162585747460
// MI455X (gfx1250) — compile-verified, matrix operand order checked by reading, not run on hardware
//
#include <hip/hip_runtime.h>
#include <hip/hip_bf16.h>
#include <math.h>

// ---------------------------------------------------------------------------
// HybridCsiNetQuantumAutoencoder for MI455X (gfx1250, wave32, WMMA).
//
// probs = | U[:, ::8] @ psi |^2  where U is the fixed 3-layer SEL circuit
// unitary. W = U[:,::8] (2048x256 complex) is built once from dec_params,
// then the batch pipeline is pure GEMM:
//   conv3x3+LeakyReLU -> h(B,2048) --WMMA--> enc(B,256) -> normalize ->
//   psi --WMMA x2--> (Wr psi, Wi psi) -> square-add -> out(B,2048).
// f16 operands / f32 accumulation via v_wmma_f32_16x16x32_f16 keeps the
// kernel HBM-bound (~128 MiB -> ~5.5us at 23.3 TB/s).
// ---------------------------------------------------------------------------

typedef __attribute__((ext_vector_type(16))) _Float16 v16h;
typedef __attribute__((ext_vector_type(8)))  float    v8f;

#define NQ    11
#define DIM   2048
#define EMB   256
#define MTILE 16

// WMMA f16 fragment swizzle helpers (ISA 7.12.2):
// A(16x32): lane = hh*16 + M,  B(32x16): lane = hh*16 + N
// slot s in v16h: K = (s>>3)*16 + hh*8 + (s&7)  =>  inverse:
//   hh = (K>>3)&1,  s = ((K>>4)<<3) | (K&7)
__device__ __forceinline__ int frag_slot(int Kin) { return ((Kin >> 4) << 3) | (Kin & 7); }
__device__ __forceinline__ int frag_hh(int Kin)   { return (Kin >> 3) & 1; }

// ---------------------------------------------------------------------------
// Kernel 0: fc_w (2048x256 f32, row-major) -> f16 fragment-major layout:
//   fcw_sw[ ((kt*16 + nt)*32 + lane)*16 + slot ]
// ---------------------------------------------------------------------------
__global__ __launch_bounds__(256) void k_swizzle_fcw(const float* __restrict__ fcw,
                                                     _Float16* __restrict__ fcw_sw) {
    int e = blockIdx.x * 256 + threadIdx.x;     // 0 .. 2048*256-1 exactly
    int k = e >> 8;                             // K index (0..2047)
    int n = e & 255;                            // N index (0..255)
    int kt = k >> 5, Kin = k & 31;
    int nt = n >> 4, nl = n & 15;
    int lane = (frag_hh(Kin) << 4) | nl;
    fcw_sw[(((kt * 16) + nt) * 32 + lane) * 16 + frag_slot(Kin)] = (_Float16)fcw[e];
}

// ---------------------------------------------------------------------------
// Kernel 1: build W = U[:, ::8].  One block per embedding column k:
// simulate the circuit on basis state e_{8k} (2048-dim complex in LDS),
// then scatter the column into Wr/Wi in GEMM2 B-fragment layout:
//   w_sw[ ((kt2*128 + nt)*32 + lane)*16 + slot ],  K = k, N = j.
// ---------------------------------------------------------------------------
__global__ __launch_bounds__(256) void k_build_w(const float* __restrict__ dec_params,
                                                 _Float16* __restrict__ wr_sw,
                                                 _Float16* __restrict__ wi_sw) {
    __shared__ float sr[DIM];
    __shared__ float si[DIM];
    __shared__ float gm[33][8];   // 33 Rot gates: re/im of m00,m01,m10,m11

    const int tid = threadIdx.x;
    const int k   = blockIdx.x;   // embedding column 0..255

    if (tid < 33) {
        int l = tid / NQ, w = tid % NQ;
        float phi = dec_params[(l * NQ + w) * 3 + 0];
        float th  = dec_params[(l * NQ + w) * 3 + 1];
        float om  = dec_params[(l * NQ + w) * 3 + 2];
        float c  = __cosf(0.5f * th), sn = __sinf(0.5f * th);
        float ap = -0.5f * (phi + om), am = -0.5f * (phi - om);
        float cp = __cosf(ap), sp = __sinf(ap);
        float cm = __cosf(am), sm = __sinf(am);
        gm[tid][0] =  cp * c;  gm[tid][1] =  sp * c;    // m00 = e^{-i(phi+om)/2} cos
        gm[tid][2] = -cm * sn; gm[tid][3] =  sm * sn;   // m01 = -conj(em) sin
        gm[tid][4] =  cm * sn; gm[tid][5] =  sm * sn;   // m10 = em sin
        gm[tid][6] =  cp * c;  gm[tid][7] = -sp * c;    // m11 = conj(ep) cos
    }
    for (int j = tid; j < DIM; j += 256) { sr[j] = 0.0f; si[j] = 0.0f; }
    __syncthreads();
    if (tid == 0) sr[k * 8] = 1.0f;   // AmplitudeEmbedding on wires 0..7 -> index k*2^3
    __syncthreads();

    for (int l = 0; l < 3; ++l) {
        // 11 single-qubit Rot gates (wire 0 = MSB -> bit position 10-w)
        for (int w = 0; w < NQ; ++w) {
            const int g = l * NQ + w;
            const float m00r = gm[g][0], m00i = gm[g][1];
            const float m01r = gm[g][2], m01i = gm[g][3];
            const float m10r = gm[g][4], m10i = gm[g][5];
            const float m11r = gm[g][6], m11i = gm[g][7];
            const int sbits  = 10 - w;
            const int stride = 1 << sbits;
            for (int p = tid; p < DIM / 2; p += 256) {
                int i0 = ((p >> sbits) << (sbits + 1)) | (p & (stride - 1));
                int i1 = i0 | stride;
                float ar = sr[i0], ai = si[i0];
                float br = sr[i1], bi = si[i1];
                sr[i0] = m00r * ar - m00i * ai + m01r * br - m01i * bi;
                si[i0] = m00r * ai + m00i * ar + m01r * bi + m01i * br;
                sr[i1] = m10r * ar - m10i * ai + m11r * br - m11i * bi;
                si[i1] = m10r * ai + m10i * ar + m11r * bi + m11i * br;
            }
            __syncthreads();
        }
        // 11 CNOTs, range r = l % (NQ-1) + 1
        const int r = (l % (NQ - 1)) + 1;
        for (int i = 0; i < NQ; ++i) {
            const int cbit = 1 << (10 - i);
            const int tbit = 1 << (10 - ((i + r) % NQ));
            for (int j = tid; j < DIM; j += 256) {
                if ((j & cbit) && !(j & tbit)) {
                    int j2 = j ^ tbit;
                    float tr = sr[j]; sr[j] = sr[j2]; sr[j2] = tr;
                    float ti = si[j]; si[j] = si[j2]; si[j2] = ti;
                }
            }
            __syncthreads();
        }
    }

    // Scatter column k of W into GEMM2 B-fragment layout (K = k, N = j)
    const int kt2  = k >> 5, Kin = k & 31;
    const int hh   = frag_hh(Kin);
    const int slot = frag_slot(Kin);
    for (int j = tid; j < DIM; j += 256) {
        int nt = j >> 4, nl = j & 15;
        int lane = (hh << 4) | nl;
        size_t off = (size_t)((kt2 * 128 + nt) * 32 + lane) * 16 + slot;
        wr_sw[off] = (_Float16)sr[j];
        wi_sw[off] = (_Float16)si[j];
    }
}

// ---------------------------------------------------------------------------
// Kernel 2: fused conv + GEMM1 + normalize + GEMM2 + |.|^2 for a 16-row tile.
// 256 threads = 8 wave32.  Dynamic LDS layout:
//   hs     : f16 A-fragments of h, 64 ktiles * 32 lanes * 16 slots = 64 KiB
//   enc_s  : f32 16x256                                           = 16 KiB
//   psi_sw : f16 A-fragments of psi, 8*32*16                      =  8 KiB
//   red    : f32 16x16 norm partials                              =  1 KiB
//   invn   : f32 16                                               = 64 B
// ---------------------------------------------------------------------------
#define LDS_HS   (MTILE * DIM * 2)            // 65536
#define LDS_ENC  (MTILE * EMB * 4)            // 16384
#define LDS_PSI  (MTILE * EMB * 2)            // 8192
#define LDS_RED  (MTILE * 16 * 4)             // 1024
#define LDS_INV  (MTILE * 4)                  // 64
#define LDS_TOTAL (LDS_HS + LDS_ENC + LDS_PSI + LDS_RED + LDS_INV)

__global__ __launch_bounds__(256) void k_fused(const float* __restrict__ x,
                                               const float* __restrict__ conv_w,
                                               const float* __restrict__ conv_b,
                                               const float* __restrict__ fc_b,
                                               const _Float16* __restrict__ fcw_sw,
                                               const _Float16* __restrict__ wr_sw,
                                               const _Float16* __restrict__ wi_sw,
                                               float* __restrict__ out) {
    extern __shared__ char smem[];
    _Float16* hs     = (_Float16*)(smem);
    float*    enc_s  = (float*)   (smem + LDS_HS);
    _Float16* psi_sw = (_Float16*)(smem + LDS_HS + LDS_ENC);
    float*    red    = (float*)   (smem + LDS_HS + LDS_ENC + LDS_PSI);
    float*    invn   = (float*)   (smem + LDS_HS + LDS_ENC + LDS_PSI + LDS_RED);
    __shared__ float cw[38];                  // 36 conv weights + 2 bias

    const int tid  = threadIdx.x;
    const int m0   = blockIdx.x * MTILE;      // first batch row of this tile
    const int wv   = tid >> 5;                // wave id 0..7
    const int lane = tid & 31;

    if (tid < 36) cw[tid] = conv_w[tid];
    if (tid < 2)  cw[36 + tid] = conv_b[tid];
    __syncthreads();

    // ---- Phase A: conv3x3 (SAME, zero-pad) + LeakyReLU(0.3) -> swizzled f16
    for (int t = tid; t < MTILE * DIM; t += 256) {
        int m   = t >> 11;
        int col = t & (DIM - 1);
        int o  = col >> 10;
        int y  = (col >> 5) & 31;
        int xx = col & 31;
        const float* xb = x + (size_t)(m0 + m) * 2048;
        float acc = cw[36 + o];
        #pragma unroll
        for (int i = 0; i < 2; ++i) {
            #pragma unroll
            for (int ky = 0; ky < 3; ++ky) {
                int yy = y + ky - 1;
                if (yy < 0 || yy > 31) continue;
                #pragma unroll
                for (int kx = 0; kx < 3; ++kx) {
                    int xc = xx + kx - 1;
                    if (xc < 0 || xc > 31) continue;
                    acc += xb[i * 1024 + yy * 32 + xc] * cw[((o * 2 + i) * 3 + ky) * 3 + kx];
                }
            }
        }
        float h = acc >= 0.0f ? acc : 0.3f * acc;
        int kt = col >> 5, Kin = col & 31;
        int la = (frag_hh(Kin) << 4) | m;
        hs[(kt * 32 + la) * 16 + frag_slot(Kin)] = (_Float16)h;
    }
    __syncthreads();

    // ---- Phase B: GEMM1  enc(16x256) = h(16x2048) @ fc_w, K-loop of 64 WMMAs
    {
        const int nt0 = wv * 2, nt1 = wv * 2 + 1;   // 16 N-tiles over 8 waves
        v8f acc0 = {}; v8f acc1 = {};
        for (int kt = 0; kt < 64; ++kt) {
            v16h a  = *(const v16h*)&hs[(kt * 32 + lane) * 16];
            v16h b0 = *(const v16h*)&fcw_sw[((kt * 16 + nt0) * 32 + lane) * 16];
            v16h b1 = *(const v16h*)&fcw_sw[((kt * 16 + nt1) * 32 + lane) * 16];
            if (kt + 1 < 64)
                __builtin_prefetch(&fcw_sw[(((kt + 1) * 16 + nt0) * 32 + lane) * 16], 0, 0);
            acc0 = __builtin_amdgcn_wmma_f32_16x16x32_f16(false, a, false, b0,
                                                          (short)0, acc0, false, false);
            acc1 = __builtin_amdgcn_wmma_f32_16x16x32_f16(false, a, false, b1,
                                                          (short)0, acc1, false, false);
        }
        // epilogue: bias, nan_to_num(nan=0, +inf=1, -inf=0), clip[0,1e7]
        #pragma unroll
        for (int v = 0; v < 8; ++v) {
            int M  = v + ((lane >> 4) << 3);
            int Nl = lane & 15;
            float e0 = acc0[v] + fc_b[nt0 * 16 + Nl];
            float e1 = acc1[v] + fc_b[nt1 * 16 + Nl];
            if (e0 != e0) e0 = 0.0f;
            if (e1 != e1) e1 = 0.0f;
            if (__builtin_isinf(e0)) e0 = e0 > 0.0f ? 1.0f : 0.0f;
            if (__builtin_isinf(e1)) e1 = e1 > 0.0f ? 1.0f : 0.0f;
            e0 = fminf(fmaxf(e0, 0.0f), 1e7f);
            e1 = fminf(fmaxf(e1, 0.0f), 1e7f);
            enc_s[M * EMB + nt0 * 16 + Nl] = e0;
            enc_s[M * EMB + nt1 * 16 + Nl] = e1;
        }
    }
    __syncthreads();

    // ---- Phase C: per-row L2 norm; psi = enc / (||enc|| + 1e-10), to f16 frags
    {
        int r = tid >> 4, seg = tid & 15;
        float p = 0.0f;
        #pragma unroll
        for (int j = 0; j < 16; ++j) {
            float v = enc_s[r * EMB + seg * 16 + j];
            p += v * v;
        }
        red[r * 16 + seg] = p;
    }
    __syncthreads();
    if (tid < MTILE) {
        float sum = 0.0f;
        #pragma unroll
        for (int j = 0; j < 16; ++j) sum += red[tid * 16 + j];
        invn[tid] = 1.0f / (sqrtf(sum) + 1e-10f);
    }
    __syncthreads();
    for (int t = tid; t < MTILE * EMB; t += 256) {
        int m = t >> 8, k = t & 255;
        float p = enc_s[m * EMB + k] * invn[m];
        int kt = k >> 5, Kin = k & 31;
        int la = (frag_hh(Kin) << 4) | m;
        psi_sw[(kt * 32 + la) * 16 + frag_slot(Kin)] = (_Float16)p;
    }
    __syncthreads();

    // ---- Phase D: GEMM2 x2 (real/imag), out = (Wr psi)^2 + (Wi psi)^2
    v16h afr[8];
    #pragma unroll
    for (int kt = 0; kt < 8; ++kt)
        afr[kt] = *(const v16h*)&psi_sw[(kt * 32 + lane) * 16];

    for (int nt = wv; nt < 128; nt += 8) {         // 128 N-tiles over 8 waves
        v8f ar = {}; v8f ai = {};
        if (nt + 8 < 128) {
            __builtin_prefetch(&wr_sw[((size_t)(nt + 8) * 32 + lane) * 16], 0, 0);
            __builtin_prefetch(&wi_sw[((size_t)(nt + 8) * 32 + lane) * 16], 0, 0);
        }
        #pragma unroll
        for (int kt = 0; kt < 8; ++kt) {
            size_t off = (size_t)((kt * 128 + nt) * 32 + lane) * 16;
            v16h br = *(const v16h*)&wr_sw[off];
            v16h bi = *(const v16h*)&wi_sw[off];
            ar = __builtin_amdgcn_wmma_f32_16x16x32_f16(false, afr[kt], false, br,
                                                        (short)0, ar, false, false);
            ai = __builtin_amdgcn_wmma_f32_16x16x32_f16(false, afr[kt], false, bi,
                                                        (short)0, ai, false, false);
        }
        #pragma unroll
        for (int v = 0; v < 8; ++v) {
            int M = v + ((lane >> 4) << 3);
            int N = nt * 16 + (lane & 15);
            float pr = ar[v], pi = ai[v];
            out[(size_t)(m0 + M) * DIM + N] = pr * pr + pi * pi;
        }
    }
}

// ---------------------------------------------------------------------------
extern "C" void kernel_launch(void* const* d_in, const int* in_sizes, int n_in,
                              void* d_out, int out_size, void* d_ws, size_t ws_size,
                              hipStream_t stream) {
    (void)in_sizes; (void)n_in; (void)out_size; (void)ws_size;
    const float* x      = (const float*)d_in[0];   // (8192,2,32,32)
    const float* conv_w = (const float*)d_in[1];   // (2,2,3,3)
    const float* conv_b = (const float*)d_in[2];   // (2,)
    const float* fc_w   = (const float*)d_in[3];   // (2048,256)
    const float* fc_b   = (const float*)d_in[4];   // (256,)
    const float* dec_p  = (const float*)d_in[5];   // (3,11,3)
    float* out = (float*)d_out;                    // (8192,2048)

    char* ws = (char*)d_ws;
    _Float16* fcw_sw = (_Float16*)(ws);                 // 1 MiB: fc_w f16 fragments
    _Float16* wr_sw  = (_Float16*)(ws + (1 << 20));     // 1 MiB: Re(W) f16 fragments
    _Float16* wi_sw  = (_Float16*)(ws + (2 << 20));     // 1 MiB: Im(W) f16 fragments

    k_swizzle_fcw<<<(DIM * EMB) / 256, 256, 0, stream>>>(fc_w, fcw_sw);
    k_build_w<<<EMB, 256, 0, stream>>>(dec_p, wr_sw, wi_sw);
    k_fused<<<8192 / MTILE, 256, LDS_TOTAL, stream>>>(x, conv_w, conv_b, fc_b,
                                                      fcw_sw, wr_sw, wi_sw, out);
}
